// SingleAttention_72782515798261
// MI455X (gfx1250) — matrix operand order repaired and checked by reading, not compiled, not run on hardware
//
#include <hip/hip_runtime.h>
#include <hip/hip_bf16.h>

typedef __attribute__((ext_vector_type(16))) __bf16 v16bf;
typedef __attribute__((ext_vector_type(8)))  float  v8f;

#define B_      64
#define S_      8192
#define D_      128
#define SCHUNK  256
#define NCHUNK  (S_ / SCHUNK)   // 32
#define NW      8               // waves per block (wave32)
#define MT      16              // rows per M tile
#define ATSTRIDE 136            // padded bf16 row stride for A staging (bank-conflict avoidance)

__device__ __forceinline__ float fast_tanh(float x) {
#if __has_builtin(__builtin_amdgcn_tanhf)
    return __builtin_amdgcn_tanhf(x);
#elif __has_builtin(__builtin_amdgcn_tanh_f32)
    return __builtin_amdgcn_tanh_f32(x);
#else
    return tanhf(x);
#endif
}

// ---------------------------------------------------------------------------
// K0: q = tanh(input @ W_dec^T + b_dec)   [B, D]  (tiny)
// ---------------------------------------------------------------------------
__global__ __launch_bounds__(D_) void qkern(const float* __restrict__ inp,
                                            const float* __restrict__ Wdec,
                                            const float* __restrict__ bdec,
                                            float* __restrict__ q) {
    __shared__ float sIn[D_];
    const int b = blockIdx.x, d = threadIdx.x;
    sIn[d] = inp[b * D_ + d];
    __syncthreads();
    float acc = bdec[d];
#pragma unroll 8
    for (int e = 0; e < D_; ++e) acc += sIn[e] * Wdec[d * D_ + e];
    q[b * D_ + d] = fast_tanh(acc);
}

// ---------------------------------------------------------------------------
// K1: fused  pc = tanh(ctx @ Wenc^T + b);  score = pc.q ;  online softmax +
//     attn-weighted context partials.  One block = (batch, 256-row S chunk).
// ---------------------------------------------------------------------------
__global__ __launch_bounds__(256) void attn_main(const float* __restrict__ ctx,
                                                 const float* __restrict__ Wenc,
                                                 const float* __restrict__ benc,
                                                 const float* __restrict__ q,
                                                 float* __restrict__ scoresOut,
                                                 float* __restrict__ partials) {
    // W_enc fragments, pre-linearized in WMMA B layout: [kt*8+nt][lane*16 + i]
    __shared__ __bf16 sB[32][32 * 16];              // 32 KB
    __shared__ __bf16 sA[NW][MT * ATSTRIDE];        // ~34 KB  (per-wave A staging)
    __shared__ float  sQ[D_], sBias[D_];
    __shared__ float  sSc[NW][MT];
    __shared__ float  sM[NW], sL[NW], sWC[NW][D_];

    const int tid  = threadIdx.x;
    const int lane = tid & 31;
    const int w    = tid >> 5;
    const int c    = blockIdx.x;    // S chunk
    const int b    = blockIdx.y;    // batch

    // ---- stage W_enc into B-fragment layout: B[k][n] = Wenc[n][k] ----------
    for (int p = tid; p < 32 * 32; p += 256) {
        const int frag = p >> 5, fl = p & 31;
        const int kt = frag >> 3, nt = frag & 7;
        const int n     = nt * 16 + (fl & 15);
        const int kbase = kt * 32 + (fl >> 4) * 16;
        const float* src = Wenc + n * D_ + kbase;   // 16 contiguous k values
        __bf16* dst = &sB[frag][fl * 16];
#pragma unroll
        for (int i = 0; i < 16; ++i) dst[i] = (__bf16)src[i];
    }
    if (tid < D_) { sQ[tid] = q[b * D_ + tid]; sBias[tid] = benc[tid]; }
    __syncthreads();

    float m_w = -3.0e38f, l_w = 0.0f;
    float wc0 = 0.f, wc1 = 0.f, wc2 = 0.f, wc3 = 0.f;  // d = lane, +32, +64, +96
    const size_t ctxBase = ((size_t)b * S_ + (size_t)c * SCHUNK) * D_;

    for (int tt = 0; tt < 2; ++tt) {
        const int t  = w + tt * NW;          // M tile 0..15 within chunk
        const int s0 = t * MT;
        const float* cTile = ctx + ctxBase + (size_t)s0 * D_;

        // ---- stage 16x128 fp32 context tile -> bf16 LDS (coalesced) -------
#pragma unroll
        for (int j = 0; j < 16; ++j) {
            const int fi   = lane + j * 32;      // float4 index in 16x32 grid
            const int row  = fi >> 5;
            const int col4 = (fi & 31) * 4;
            const float4 cv = *(const float4*)(cTile + row * D_ + col4);
            __bf16* dst = &sA[w][row * ATSTRIDE + col4];
            dst[0] = (__bf16)cv.x; dst[1] = (__bf16)cv.y;
            dst[2] = (__bf16)cv.z; dst[3] = (__bf16)cv.w;
        }
        asm volatile("s_wait_dscnt 0" ::: "memory");   // wave-local LDS publish

        // ---- A fragments (16x32, contiguous k = 16*(lane/16) + s, same as sB) ----------------------
        union { v16bf v; unsigned u[8]; } af[4];
        const int rowA = lane & 15;
        const int hi8  = (lane & 16) ? 8 : 0;
#pragma unroll
        for (int kt = 0; kt < 4; ++kt) {
#pragma unroll
            for (int v = 0; v < 8; ++v) {
                const int k = kt * 32 + hi8 * 2 + v * 2;
                af[kt].u[v] =
                    *(const unsigned*)((const __bf16*)&sA[w][0] + rowA * ATSTRIDE + k);
            }
        }

        // ---- GEMM: acc[nt] = bias + A(16x128) x B(128x16nt) ---------------
        v8f acc[8];
#pragma unroll
        for (int nt = 0; nt < 8; ++nt) {
            const float bias = sBias[nt * 16 + (lane & 15)];
#pragma unroll
            for (int j = 0; j < 8; ++j) acc[nt][j] = bias;
        }
#pragma unroll
        for (int nt = 0; nt < 8; ++nt) {
#pragma unroll
            for (int kt = 0; kt < 4; ++kt) {
                const v16bf bv = *(const v16bf*)&sB[kt * 8 + nt][lane * 16];
                acc[nt] = __builtin_amdgcn_wmma_f32_16x16x32_bf16(
                    false, af[kt].v, false, bv, (short)0, acc[nt], false, false);
            }
        }

        // ---- score[row] = sum_d tanh(acc) * q[d]  (cross-lane reduce) -----
        float part[8];
#pragma unroll
        for (int j = 0; j < 8; ++j) part[j] = 0.f;
#pragma unroll
        for (int nt = 0; nt < 8; ++nt) {
            const float qv = sQ[nt * 16 + (lane & 15)];
#pragma unroll
            for (int j = 0; j < 8; ++j) part[j] += fast_tanh(acc[nt][j]) * qv;
        }
#pragma unroll
        for (int j = 0; j < 8; ++j) {
            float p = part[j];
            p += __shfl_xor(p, 1, 16);
            p += __shfl_xor(p, 2, 16);
            p += __shfl_xor(p, 4, 16);
            p += __shfl_xor(p, 8, 16);
            part[j] = p;
        }
        if ((lane & 15) == 0) {
            const int mbase = (lane >> 4) * 8;   // lane0 -> rows 0..7, lane16 -> 8..15
#pragma unroll
            for (int j = 0; j < 8; ++j) {
                sSc[w][mbase + j] = part[j];
                scoresOut[(size_t)b * S_ + c * SCHUNK + s0 + mbase + j] = part[j];
            }
        }
        asm volatile("s_wait_dscnt 0" ::: "memory");

        // ---- online softmax + weighted raw-context accumulation -----------
        float tmax = sSc[w][0];
#pragma unroll
        for (int s = 1; s < 16; ++s) tmax = fmaxf(tmax, sSc[w][s]);
        const float m_new = fmaxf(m_w, tmax);
        const float scale = __expf(m_w - m_new);
        l_w *= scale; wc0 *= scale; wc1 *= scale; wc2 *= scale; wc3 *= scale;
#pragma unroll 4
        for (int s = 0; s < 16; ++s) {
            const float e = __expf(sSc[w][s] - m_new);
            l_w += e;
            const float* crow = cTile + s * D_;     // L0/L2-hot re-read, fp32
            wc0 += e * crow[lane];
            wc1 += e * crow[lane + 32];
            wc2 += e * crow[lane + 64];
            wc3 += e * crow[lane + 96];
        }
        m_w = m_new;
    }

    // ---- combine 8 waves -> one per-chunk partial (m, l, wc[128]) ---------
    if (lane == 0) { sM[w] = m_w; sL[w] = l_w; }
    sWC[w][lane]      = wc0;
    sWC[w][lane + 32] = wc1;
    sWC[w][lane + 64] = wc2;
    sWC[w][lane + 96] = wc3;
    __syncthreads();
    if (tid < D_) {
        float mg = sM[0];
#pragma unroll
        for (int wv = 1; wv < NW; ++wv) mg = fmaxf(mg, sM[wv]);
        float lg = 0.f, wcg = 0.f;
#pragma unroll
        for (int wv = 0; wv < NW; ++wv) {
            const float f = __expf(sM[wv] - mg);
            lg  += sL[wv] * f;
            wcg += sWC[wv][tid] * f;
        }
        float* rec = partials + ((size_t)b * NCHUNK + c) * (D_ + 2);
        if (tid == 0) { rec[0] = mg; rec[1] = lg; }
        rec[2 + tid] = wcg;
    }
}

// ---------------------------------------------------------------------------
// K2: per-batch combine of 32 chunk partials + output projection (tiny)
// ---------------------------------------------------------------------------
__global__ __launch_bounds__(D_) void finalize(const float* __restrict__ q,
                                               const float* __restrict__ Wout,
                                               const float* __restrict__ partials,
                                               float* __restrict__ h,
                                               float* __restrict__ ml) {
    __shared__ float sCat[2 * D_];
    __shared__ float sMC[NCHUNK], sLC[NCHUNK];
    const int b = blockIdx.x, d = threadIdx.x;
    const float* pb = partials + (size_t)b * NCHUNK * (D_ + 2);
    if (d < NCHUNK) { sMC[d] = pb[d * (D_ + 2)]; sLC[d] = pb[d * (D_ + 2) + 1]; }
    __syncthreads();
    float mg = sMC[0];
#pragma unroll
    for (int cc = 1; cc < NCHUNK; ++cc) mg = fmaxf(mg, sMC[cc]);
    float lg = 0.f, wc = 0.f;
#pragma unroll 4
    for (int cc = 0; cc < NCHUNK; ++cc) {
        const float f = __expf(sMC[cc] - mg);
        lg += sLC[cc] * f;
        wc += pb[cc * (D_ + 2) + 2 + d] * f;
    }
    if (d == 0) { ml[b * 2] = mg; ml[b * 2 + 1] = lg; }
    sCat[d]      = wc / lg;          // normalized weighted context
    sCat[D_ + d] = q[b * D_ + d];
    __syncthreads();
    float acc = 0.f;
#pragma unroll 8
    for (int j = 0; j < 2 * D_; ++j) acc += sCat[j] * Wout[d * 2 * D_ + j];
    h[b * D_ + d] = fast_tanh(acc);
}

// ---------------------------------------------------------------------------
// K3: attn = exp(score - m_g) / l_g   (rescale raw scores in-place in d_out)
// ---------------------------------------------------------------------------
__global__ __launch_bounds__(256) void attn_norm(float* __restrict__ attn,
                                                 const float* __restrict__ ml) {
    const size_t idx = (size_t)blockIdx.x * blockDim.x + threadIdx.x;
    const int b = (int)(idx / S_);
    attn[idx] = __expf(attn[idx] - ml[b * 2]) / ml[b * 2 + 1];
}

// ---------------------------------------------------------------------------
extern "C" void kernel_launch(void* const* d_in, const int* in_sizes, int n_in,
                              void* d_out, int out_size, void* d_ws, size_t ws_size,
                              hipStream_t stream) {
    const float* inp  = (const float*)d_in[0];
    const float* ctx  = (const float*)d_in[1];
    const float* Wenc = (const float*)d_in[2];
    const float* benc = (const float*)d_in[3];
    const float* Wdec = (const float*)d_in[4];
    const float* bdec = (const float*)d_in[5];
    const float* Wout = (const float*)d_in[6];

    float* h    = (float*)d_out;            // [B, D]
    float* attn = (float*)d_out + B_ * D_;  // [B, S]

    float* q    = (float*)d_ws;                         // B*D
    float* part = q + B_ * D_;                          // B*NCHUNK*(D+2)
    float* ml   = part + (size_t)B_ * NCHUNK * (D_ + 2); // B*2

    qkern    <<<dim3(B_),         dim3(D_),  0, stream>>>(inp, Wdec, bdec, q);
    attn_main<<<dim3(NCHUNK, B_), dim3(256), 0, stream>>>(ctx, Wenc, benc, q, attn, part);
    finalize <<<dim3(B_),         dim3(D_),  0, stream>>>(q, Wout, part, h, ml);
    attn_norm<<<dim3((B_ * S_) / 256), dim3(256), 0, stream>>>(attn, ml);
}
